// Attention_83829171683526
// MI455X (gfx1250) — matrix operand order repaired and checked by reading, not compiled, not run on hardware
//
#include <hip/hip_runtime.h>

#define DIM    1024
#define NHEADS 16
#define HDIM   64
#define BB     4
#define NN     2048
#define MROWS  (BB * NN)   // 8192

typedef __attribute__((ext_vector_type(16))) __bf16 v16bf;
typedef __attribute__((ext_vector_type(8)))  __bf16 v8bf;
typedef __attribute__((ext_vector_type(8)))  float  v8f;
typedef __attribute__((ext_vector_type(4)))  float  v4f;

__device__ __forceinline__ v16bf pack16(v8bf a, v8bf b) {
  return __builtin_shufflevector(a, b, 0,1,2,3,4,5,6,7,8,9,10,11,12,13,14,15);
}

__device__ __forceinline__ v8bf cvt8(v4f a, v4f b) {
  v8bf r;
  r[0]=(__bf16)a[0]; r[1]=(__bf16)a[1]; r[2]=(__bf16)a[2]; r[3]=(__bf16)a[3];
  r[4]=(__bf16)b[0]; r[5]=(__bf16)b[1]; r[6]=(__bf16)b[2]; r[7]=(__bf16)b[3];
  return r;
}

// --- WMMA operand loaders ------------------------------------------------
// A operand (16xK, row-major source): lane L -> row (L&15);
// elements 0..15 hold contiguous K = k0 + g*16 .. +15
__device__ __forceinline__ v16bf loadA_f32(const float* __restrict__ base, int ld,
                                           int row0, int k0, int lane) {
  int r = lane & 15, g = lane >> 4;
  const float* p = base + (size_t)(row0 + r) * ld + k0 + g * 16;
  v4f a0 = *(const v4f*)(p);
  v4f a1 = *(const v4f*)(p + 4);
  v4f b0 = *(const v4f*)(p + 8);
  v4f b1 = *(const v4f*)(p + 12);
  return pack16(cvt8(a0, a1), cvt8(b0, b1));
}

__device__ __forceinline__ v16bf loadA_bf16(const __bf16* __restrict__ base, int ld,
                                            int row0, int k0, int lane) {
  int r = lane & 15, g = lane >> 4;
  const __bf16* p = base + (size_t)(row0 + r) * ld + k0 + g * 16;
  v8bf a = *(const v8bf*)(p);
  v8bf b = *(const v8bf*)(p + 8);
  return pack16(a, b);
}

// B operand (Kx16 as 16 row-major "columns"): lane L -> column (L&15);
// elements 0..15 hold contiguous K = k0 + g*16 .. +15
__device__ __forceinline__ v16bf loadB_f32(const float* __restrict__ base, int ld,
                                           int col0, int k0, int lane) {
  int r = lane & 15, g = lane >> 4;
  const float* p = base + (size_t)(col0 + r) * ld + k0 + g * 16;
  v4f a0 = *(const v4f*)(p);
  v4f a1 = *(const v4f*)(p + 4);
  v4f b0 = *(const v4f*)(p + 8);
  v4f b1 = *(const v4f*)(p + 12);
  return pack16(cvt8(a0, a1), cvt8(b0, b1));
}

__device__ __forceinline__ v16bf loadB_bf16(const __bf16* __restrict__ base, int ld,
                                            int col0, int k0, int lane) {
  int r = lane & 15, g = lane >> 4;
  const __bf16* p = base + (size_t)(col0 + r) * ld + k0 + g * 16;
  v8bf a = *(const v8bf*)(p);
  v8bf b = *(const v8bf*)(p + 8);
  return pack16(a, b);
}

__device__ __forceinline__ v8f wmma_bf16(v16bf a, v16bf b, v8f c) {
  return __builtin_amdgcn_wmma_f32_16x16x32_bf16(false, a, false, b, (short)0, c,
                                                 false, false);
}

// ========================= Kernel A: QKV GEMM ============================
// y = x @ W^T; WHICH=0 -> Q (scaled, (B,H,N,hd)), 1 -> K, 2 -> V^T (B,H,hd,N).
// Each wave owns a 16x32 output tile: one A load feeds two WMMAs.
template <int WHICH>
__global__ void __launch_bounds__(256) qkv_kernel(
    const float* __restrict__ x, const float* __restrict__ W,
    __bf16* __restrict__ dst) {
  int lane = threadIdx.x & 31;
  int wave = threadIdx.x >> 5;
  int m0   = blockIdx.x * 16;
  int col0 = blockIdx.y * 256 + wave * 32;

  v8f acc[2] = {{}, {}};
#pragma unroll 2
  for (int k0 = 0; k0 < DIM; k0 += 32) {
    // prefetch next K-step operands (global_prefetch_b8; L2-resident reuse)
    __builtin_prefetch(x + (size_t)(m0 + (lane & 15)) * DIM + k0 + 64, 0, 3);
    __builtin_prefetch(W + (size_t)(col0 + (lane & 15)) * DIM + k0 + 64, 0, 3);
    v16bf a  = loadA_f32(x, DIM, m0, k0, lane);
    v16bf b0 = loadB_f32(W, DIM, col0, k0, lane);
    v16bf b1 = loadB_f32(W, DIM, col0 + 16, k0, lane);
    acc[0] = wmma_bf16(a, b0, acc[0]);
    acc[1] = wmma_bf16(a, b1, acc[1]);
  }

  int r15 = lane & 15, g = lane >> 4;
#pragma unroll
  for (int t = 0; t < 2; ++t) {
    int col = col0 + t * 16 + r15;
    int h = col >> 6, d = col & 63;
#pragma unroll
    for (int r = 0; r < 8; ++r) {
      int m = m0 + r + g * 8;
      int b = m >> 11;           // / NN
      int n = m & (NN - 1);
      size_t bh = (size_t)(b * NHEADS + h);
      float v = acc[t][r];
      if (WHICH == 0)
        dst[(bh * NN + n) * HDIM + d] = (__bf16)(v * 0.125f);  // HDIM^-0.5
      else if (WHICH == 1)
        dst[(bh * NN + n) * HDIM + d] = (__bf16)v;
      else
        dst[(bh * HDIM + d) * NN + n] = (__bf16)v;
    }
  }
}

// ================== Kernel B: flash attention (per 16-q wave) ============
// S^T = K @ Q^T per 16-key tile (softmax reductions stay in-lane), online
// softmax, then O^T += V^T @ P^T with the P^T B-operand built in registers.
__global__ void __launch_bounds__(128) attn_kernel(
    const __bf16* __restrict__ Qb, const __bf16* __restrict__ Kb,
    const __bf16* __restrict__ Vt, const int* __restrict__ mask,
    __bf16* __restrict__ Ob) {
  int lane = threadIdx.x & 31;
  int wave = threadIdx.x >> 5;
  int gw = blockIdx.x * 4 + wave;
  int qt = gw & ((NN / 16) - 1);   // query tile 0..127
  int bh = gw >> 7;                // 0..63
  int b = bh >> 4, h = bh & 15;

  const __bf16* Qh = Qb + (size_t)bh * NN * HDIM;
  const __bf16* Kh = Kb + (size_t)bh * NN * HDIM;
  const __bf16* Vh = Vt + (size_t)bh * HDIM * NN;
  const int*    mb = mask + (size_t)b * NN * NN;

  int r15 = lane & 15, g = lane >> 4;
  int q = qt * 16 + r15;  // query column owned by this lane in S^T / O^T

  // Q rows as B-operand columns, cached for the whole key loop (hd = 64)
  v16bf qop0 = loadB_bf16(Qh, HDIM, qt * 16, 0, lane);
  v16bf qop1 = loadB_bf16(Qh, HDIM, qt * 16, 32, lane);

  v8f o0 = {}, o1 = {}, o2 = {}, o3 = {};
  float mrow = -1e30f, lrow = 0.f;

  for (int key0 = 0; key0 < NN; key0 += 32) {
    // prefetch next step's K rows and mask line
    __builtin_prefetch(Kh + (size_t)(key0 + 32 + r15) * HDIM, 0, 3);
    __builtin_prefetch(mb + (size_t)q * NN + key0 + 32, 0, 3);

    // S^T tiles for keys [key0,key0+16) and [key0+16,key0+32)
    v8f s0 = {}, s1 = {};
    s0 = wmma_bf16(loadA_bf16(Kh, HDIM, key0, 0, lane), qop0, s0);
    s0 = wmma_bf16(loadA_bf16(Kh, HDIM, key0, 32, lane), qop1, s0);
    s1 = wmma_bf16(loadA_bf16(Kh, HDIM, key0 + 16, 0, lane), qop0, s1);
    s1 = wmma_bf16(loadA_bf16(Kh, HDIM, key0 + 16, 32, lane), qop1, s1);

    // mask: this lane covers query q, keys key0 + g*8 + r (+16 for tile 1)
    const int* mp = mb + (size_t)q * NN + key0 + g * 8;
#pragma unroll
    for (int r = 0; r < 8; ++r) {
      s0[r] = mp[r]      ? s0[r] : -1e30f;
      s1[r] = mp[16 + r] ? s1[r] : -1e30f;
    }

    // online softmax (per query = per lane; combine the two key halves)
    float tmax = -1e30f;
#pragma unroll
    for (int r = 0; r < 8; ++r) tmax = fmaxf(tmax, fmaxf(s0[r], s1[r]));
    tmax = fmaxf(tmax, __shfl_xor(tmax, 16, 32));
    float mnew = fmaxf(mrow, tmax);
    float f = __expf(mrow - mnew);

    float p0[8], p1[8], tsum = 0.f;
#pragma unroll
    for (int r = 0; r < 8; ++r) {
      p0[r] = __expf(s0[r] - mnew);
      p1[r] = __expf(s1[r] - mnew);
      tsum += p0[r] + p1[r];
    }
    tsum += __shfl_xor(tsum, 16, 32);
    lrow = lrow * f + tsum;
    mrow = mnew;
#pragma unroll
    for (int r = 0; r < 8; ++r) { o0[r] *= f; o1[r] *= f; o2[r] *= f; o3[r] *= f; }

    // Build P^T B-operand: lane = query col, element j -> key j + g*16.
    // Own half supplies keys g*8+r of each tile; partner (xor 16) the rest.
    v16bf pb;
#pragma unroll
    for (int r = 0; r < 8; ++r) {
      float x0 = __shfl_xor(p0[r], 16, 32);
      float x1 = __shfl_xor(p1[r], 16, 32);
      pb[r]     = (__bf16)(g ? x1 : p0[r]);
      pb[r + 8] = (__bf16)(g ? p1[r] : x0);
    }

    // O^T(d,q) += V^T(d, keys) @ P^T(keys, q), K = 32
    o0 = wmma_bf16(loadA_bf16(Vh, NN, 0, key0, lane), pb, o0);
    o1 = wmma_bf16(loadA_bf16(Vh, NN, 16, key0, lane), pb, o1);
    o2 = wmma_bf16(loadA_bf16(Vh, NN, 32, key0, lane), pb, o2);
    o3 = wmma_bf16(loadA_bf16(Vh, NN, 48, key0, lane), pb, o3);
  }

  float inv = 1.0f / lrow;
  __bf16* op = Ob + ((size_t)b * NN + qt * 16 + r15) * DIM + h * HDIM;
#pragma unroll
  for (int r = 0; r < 8; ++r) {
    op[r + g * 8]      = (__bf16)(o0[r] * inv);
    op[16 + r + g * 8] = (__bf16)(o1[r] * inv);
    op[32 + r + g * 8] = (__bf16)(o2[r] * inv);
    op[48 + r + g * 8] = (__bf16)(o3[r] * inv);
  }
}

// ===================== Kernel C: output projection =======================
__global__ void __launch_bounds__(256) proj_kernel(
    const __bf16* __restrict__ Ob, const float* __restrict__ wp,
    const float* __restrict__ bp, float* __restrict__ out) {
  int lane = threadIdx.x & 31;
  int wave = threadIdx.x >> 5;
  int m0   = blockIdx.x * 16;
  int col0 = blockIdx.y * 256 + wave * 32;

  v8f acc[2] = {{}, {}};
#pragma unroll 2
  for (int k0 = 0; k0 < DIM; k0 += 32) {
    __builtin_prefetch(Ob + (size_t)(m0 + (lane & 15)) * DIM + k0 + 64, 0, 3);
    __builtin_prefetch(wp + (size_t)(col0 + (lane & 15)) * DIM + k0 + 64, 0, 3);
    v16bf a  = loadA_bf16(Ob, DIM, m0, k0, lane);
    v16bf b0 = loadB_f32(wp, DIM, col0, k0, lane);
    v16bf b1 = loadB_f32(wp, DIM, col0 + 16, k0, lane);
    acc[0] = wmma_bf16(a, b0, acc[0]);
    acc[1] = wmma_bf16(a, b1, acc[1]);
  }

  int r15 = lane & 15, g = lane >> 4;
#pragma unroll
  for (int t = 0; t < 2; ++t) {
    int col = col0 + t * 16 + r15;
    float bias = bp[col];
#pragma unroll
    for (int r = 0; r < 8; ++r) {
      int m = m0 + r + g * 8;
      out[(size_t)m * DIM + col] = acc[t][r] + bias;
    }
  }
}

extern "C" void kernel_launch(void* const* d_in, const int* in_sizes, int n_in,
                              void* d_out, int out_size, void* d_ws, size_t ws_size,
                              hipStream_t stream) {
  const float* x    = (const float*)d_in[0];
  const int*   mask = (const int*)d_in[1];
  const float* wq   = (const float*)d_in[2];
  const float* wk   = (const float*)d_in[3];
  const float* wv   = (const float*)d_in[4];
  const float* wp   = (const float*)d_in[5];
  const float* bp   = (const float*)d_in[6];
  float* out = (float*)d_out;

  size_t elems = (size_t)BB * NN * DIM;  // 8M elements per tensor
  __bf16* Qb = (__bf16*)d_ws;            // 16 MB
  __bf16* Kb = Qb + elems;               // 16 MB
  __bf16* Vt = Kb + elems;               // 16 MB (transposed V)
  __bf16* Ob = Vt + elems;               // 16 MB (attention output, (B,N,DIM))

  dim3 ggemm(MROWS / 16, DIM / 256);
  qkv_kernel<0><<<ggemm, 256, 0, stream>>>(x, wq, Qb);
  qkv_kernel<1><<<ggemm, 256, 0, stream>>>(x, wk, Kb);
  qkv_kernel<2><<<ggemm, 256, 0, stream>>>(x, wv, Vt);
  attn_kernel<<<dim3((BB * NHEADS * (NN / 16)) / 4), 128, 0, stream>>>(
      Qb, Kb, Vt, mask, Ob);
  proj_kernel<<<ggemm, 256, 0, stream>>>(Ob, wp, bp, out);
}
